// GlobalResidualVLAD_58239756533929
// MI455X (gfx1250) — compile-verified, matrix operand order checked by reading, not run on hardware
//
#include <hip/hip_runtime.h>
#include <hip/hip_bf16.h>

// Problem constants (from reference: B,D,K,P)
#define Bn 65536
#define Dn 512
#define Kn 128
#define Pn 512

// d_ws layout (bytes)
#define CT_OFF   0        // bf16 C^T, D x K  -> 131072 B
#define WBF_OFF  131072   // bf16 W,  P x D   -> 524288 B
#define C2_OFF   655360   // f32 ||C_k||^2, K -> 512 B
#define INV_OFF  655872   // f32 inv-norm per row, B -> 262144 B  (total 918016 B)

// LDS layout: [ C^T bf16 128KB ][ per-wave: abf 4KB | S(f32,8KB)/R(bf16,16KB) aliased ] x 8
#define LDS_CT_BYTES 131072
#define ABF_BYTES    4096
#define WAVE_REGION  20480
#define LDS_TOTAL    (LDS_CT_BYTES + 8 * WAVE_REGION)   // 294912 <= 320KB

typedef __attribute__((ext_vector_type(2)))  float  v2f;
typedef __attribute__((ext_vector_type(8)))  float  v8f;
typedef __attribute__((ext_vector_type(8)))  __bf16 v8bf;
typedef __attribute__((ext_vector_type(16))) __bf16 v16bf;

union U16B { uint4 u; v8bf b; };

static __device__ inline v16bf mk16(v8bf lo, v8bf hi) {
  v16bf r;
#pragma unroll
  for (int j = 0; j < 8; ++j) { r[j] = lo[j]; r[8 + j] = hi[j]; }
  return r;
}
static __device__ inline v8f zero8() {
  v8f r;
#pragma unroll
  for (int j = 0; j < 8; ++j) r[j] = 0.0f;
  return r;
}

// ---------------------------------------------------------------------------
// Kernel 1: precompute bf16 C^T (D x K), bf16 W (P x D), and c2[k] = ||C_k||^2
// ---------------------------------------------------------------------------
__global__ __launch_bounds__(256) void vlad_prep(const float* __restrict__ C,
                                                 const float* __restrict__ W,
                                                 char* __restrict__ ws) {
  const int i = blockIdx.x * 256 + threadIdx.x;  // grid covers P*D = 262144
  __bf16* Ct = (__bf16*)(ws + CT_OFF);
  __bf16* Wb = (__bf16*)(ws + WBF_OFF);
  if (i < Pn * Dn) Wb[i] = (__bf16)W[i];
  if (i < Dn * Kn) {
    const int d = i >> 7;        // / Kn
    const int k = i & (Kn - 1);  // % Kn
    Ct[i] = (__bf16)C[(size_t)k * Dn + d];
  }
  if (i < Kn) {
    const float4* cr = (const float4*)(C + (size_t)i * Dn);
    float s = 0.f;
#pragma unroll 4
    for (int j = 0; j < Dn / 4; ++j) {
      float4 v = cr[j];
      s += v.x * v.x + v.y * v.y + v.z * v.z + v.w * v.w;
    }
    ((float*)(ws + C2_OFF))[i] = s;
  }
}

// ---------------------------------------------------------------------------
// Kernel 2: main fused VLAD kernel.
// 256 threads = 8 waves; each wave owns a 16-row tile of z. 512 blocks.
//   Phase 1: S = z16 @ C^T   (fp32 WMMA 16x16x4, precision-critical)
//   Phase 2: softmax rows -> a (f32 to d_out, bf16 to LDS)
//   Phase 3: R = z - a @ C   (bf16 WMMA 16x16x32, C^T from LDS) -> bf16 LDS
//   Phase 4: dvec = R @ W^T + b (bf16 WMMA), store unnormalized + row invnorm
// ---------------------------------------------------------------------------
__global__ __launch_bounds__(256) void vlad_main(const float* __restrict__ z,
                                                 const float* __restrict__ C,
                                                 const float* __restrict__ alpha_raw,
                                                 const float* __restrict__ bvec,
                                                 char* __restrict__ ws,
                                                 float* __restrict__ out) {
  extern __shared__ char smem[];
  const int tid  = threadIdx.x;
  const int lane = tid & 31;
  const int wave = tid >> 5;
  const int l15  = lane & 15;
  const int h    = lane >> 4;           // half-wave select (K-chunk offset)
  const int rw   = blockIdx.x * 128 + wave * 16;  // first global row of tile

  // ---- stage bf16 C^T (D x K) into LDS, cooperatively ----
  {
    const uint4* src = (const uint4*)(ws + CT_OFF);
    uint4* dst = (uint4*)smem;
#pragma unroll 4
    for (int i = tid; i < LDS_CT_BYTES / 16; i += 256) dst[i] = src[i];
  }
  __syncthreads();

  char*   waveBase = smem + LDS_CT_BYTES + wave * WAVE_REGION;
  __bf16* abf = (__bf16*)waveBase;          // 16 x 128 bf16 soft-assignments
  char*   buf = waveBase + ABF_BYTES;       // aliased: S (f32 16x128) then R (bf16 16x512)
  float*  Sf  = (float*)buf;
  __bf16* Rw  = (__bf16*)buf;

  // ---- z2 partial: lane covers row l15, half h of D ----
  float z2p = 0.f;
  {
    const float4* zr = (const float4*)(z + (size_t)(rw + l15) * Dn + h * 256);
#pragma unroll 8
    for (int i = 0; i < 64; ++i) {
      float4 v = zr[i];
      z2p += v.x * v.x + v.y * v.y + v.z * v.z + v.w * v.w;
    }
    z2p += __shfl_xor(z2p, 16, 32);  // lanes 0-15 now hold full ||z_row||^2
  }

  // ---- Phase 1: S = z16 @ C^T via V_WMMA_F32_16X16X4_F32 ----
  v8f S[8];
#pragma unroll
  for (int t = 0; t < 8; ++t) S[t] = zero8();
  {
    const float* zrow = z + (size_t)(rw + l15) * Dn;
#pragma unroll 1
    for (int k0 = 0; k0 < Dn; k0 += 4) {
      const int koff = k0 + 2 * h;  // A layout: lanes0-15 K={k0,k0+1}; lanes16-31 K={k0+2,k0+3}
      v2f a;
      { float2 t2 = *(const float2*)(zrow + koff); a.x = t2.x; a.y = t2.y; }
#pragma unroll
      for (int t = 0; t < 8; ++t) {
        const int code = t * 16 + l15;  // B: col-per-lane; B[k][n] = C[n][k]
        v2f bf;
        { float2 t2 = *(const float2*)(C + (size_t)code * Dn + koff); bf.x = t2.x; bf.y = t2.y; }
        S[t] = __builtin_amdgcn_wmma_f32_16x16x4_f32(false, a, false, bf, (short)0,
                                                     S[t], false, false);
      }
    }
  }
  // spill raw S (f32) to LDS in row-major for the softmax pass
#pragma unroll
  for (int t = 0; t < 8; ++t) {
#pragma unroll
    for (int v = 0; v < 8; ++v) {
      const int m = v + 8 * h;  // C/D layout: lanes0-15 -> M=v, lanes16-31 -> M=v+8
      Sf[m * Kn + t * 16 + l15] = S[t][v];
    }
  }

  // ---- Phase 2: row softmax (lanes 0-15, one row each) ----
  {
    const float alpha = log1pf(__expf(alpha_raw[0])) + 1e-6f;  // softplus + eps
    const float* c2 = (const float*)(ws + C2_OFF);
    if (lane < 16) {
      const int r = lane;
      float* row = Sf + r * Kn;
      float mx = -3.4e38f;
#pragma unroll 4
      for (int k = 0; k < Kn; ++k) {
        float d2 = z2p + c2[k] - 2.0f * row[k];
        d2 = fmaxf(d2, 0.0f);
        const float lg = -alpha * d2;
        row[k] = lg;
        mx = fmaxf(mx, lg);
      }
      float sum = 0.f;
#pragma unroll 4
      for (int k = 0; k < Kn; ++k) {
        const float e = __expf(row[k] - mx);
        row[k] = e;
        sum += e;
      }
      const float inv = 1.0f / sum;
      float* aOut = out + (size_t)Bn * Pn + (size_t)(rw + r) * Kn;
#pragma unroll 4
      for (int k = 0; k < Kn; ++k) {
        const float av = row[k] * inv;
        abf[r * Kn + k] = (__bf16)av;  // bf16 copy for the a@C WMMA
        aOut[k] = av;                  // exact f32 'a' output
      }
    }
  }
  // (DS ops are in-order per wave: abf written above is visible to all lanes below)

  // ---- Phase 3: R = z - a @ C via V_WMMA_F32_16X16X32_BF16; R -> bf16 LDS ----
#pragma unroll 1
  for (int t = 0; t < 32; ++t) {
    v8f acc = zero8();
#pragma unroll
    for (int ks = 0; ks < 4; ++ks) {
      const int k0 = ks * 32;
      U16B alo, ahi, blo, bhi;
      // A = a (16x128): row-per-lane, K chunks [8h..8h+7], [16+8h..16+8h+7]
      alo.u = *(const uint4*)((const char*)abf + (l15 * Kn + k0 + 8 * h) * 2);
      ahi.u = *(const uint4*)((const char*)abf + (l15 * Kn + k0 + 16 + 8 * h) * 2);
      // B = C (128x512 tile): col-per-lane; B[k][n] = C[k][16t+n] = Ct[16t+n][k]
      const char* ctb = smem + (size_t)(t * 16 + l15) * (Kn * 2);
      blo.u = *(const uint4*)(ctb + (k0 + 8 * h) * 2);
      bhi.u = *(const uint4*)(ctb + (k0 + 16 + 8 * h) * 2);
      acc = __builtin_amdgcn_wmma_f32_16x16x32_bf16(false, mk16(alo.b, ahi.b),
                                                    false, mk16(blo.b, bhi.b),
                                                    (short)0, acc, false, false);
    }
    const int d = t * 16 + l15;
#pragma unroll
    for (int v = 0; v < 8; ++v) {
      const int m = v + 8 * h;
      const float r = z[(size_t)(rw + m) * Dn + d] - acc[v];  // R = z - a@C
      Rw[m * Dn + d] = (__bf16)r;  // overwrites dead S region; abf untouched
    }
  }

  // ---- Phase 4: dvec = R @ W^T + b, chunked over P (8 chunks x 64 cols) ----
  float ssq[8];
#pragma unroll
  for (int v = 0; v < 8; ++v) ssq[v] = 0.f;
  const char* wsWb = ws + WBF_OFF;
#pragma unroll 1
  for (int pc = 0; pc < 8; ++pc) {
    v8f acc4[4];
#pragma unroll
    for (int t = 0; t < 4; ++t) acc4[t] = zero8();
#pragma unroll 1
    for (int k0 = 0; k0 < Dn; k0 += 32) {
      U16B alo, ahi;
      const char* rb = (const char*)Rw + l15 * (Dn * 2);  // A = R, row-per-lane
      alo.u = *(const uint4*)(rb + (k0 + 8 * h) * 2);
      ahi.u = *(const uint4*)(rb + (k0 + 16 + 8 * h) * 2);
      const v16bf af = mk16(alo.b, ahi.b);
#pragma unroll
      for (int t = 0; t < 4; ++t) {
        const int p = pc * 64 + t * 16 + l15;  // B[k][n] = W^T[k][p] = W[p][k]
        const char* wb = wsWb + (size_t)p * (Dn * 2);
        U16B wlo, whi;
        wlo.u = *(const uint4*)(wb + (k0 + 8 * h) * 2);
        whi.u = *(const uint4*)(wb + (k0 + 16 + 8 * h) * 2);
        acc4[t] = __builtin_amdgcn_wmma_f32_16x16x32_bf16(false, af, false,
                                                          mk16(wlo.b, whi.b),
                                                          (short)0, acc4[t], false, false);
      }
    }
#pragma unroll
    for (int t = 0; t < 4; ++t) {
      const int p = pc * 64 + t * 16 + l15;
      const float bb = bvec[p];
#pragma unroll
      for (int v = 0; v < 8; ++v) {
        const int m = v + 8 * h;
        const float val = acc4[t][v] + bb;
        out[(size_t)(rw + m) * Pn + p] = val;  // unnormalized dvec
        ssq[v] += val * val;
      }
    }
  }

  // ---- per-row inverse norm -> d_ws (cols of row m live in 16 lanes of a half-wave) ----
  {
    float* invn = (float*)(ws + INV_OFF);
#pragma unroll
    for (int v = 0; v < 8; ++v) {
      float s = ssq[v];
      s += __shfl_xor(s, 1, 32);
      s += __shfl_xor(s, 2, 32);
      s += __shfl_xor(s, 4, 32);
      s += __shfl_xor(s, 8, 32);
      if (l15 == 0) invn[rw + v + 8 * h] = 1.0f / fmaxf(sqrtf(s), 1e-12f);
    }
  }
}

// ---------------------------------------------------------------------------
// Kernel 3: scale dvec rows by the precomputed inverse norms (float4 sweep).
// ---------------------------------------------------------------------------
__global__ __launch_bounds__(256) void vlad_norm(float* __restrict__ out,
                                                 const float* __restrict__ invn) {
  const int i = blockIdx.x * 256 + threadIdx.x;  // over B * P/4
  const int row = i >> 7;                        // P/4 = 128 float4 per row
  const float s = invn[row];
  float4* o = (float4*)out;
  float4 v = o[i];
  v.x *= s; v.y *= s; v.z *= s; v.w *= s;
  o[i] = v;
}

extern "C" void kernel_launch(void* const* d_in, const int* in_sizes, int n_in,
                              void* d_out, int out_size, void* d_ws, size_t ws_size,
                              hipStream_t stream) {
  const float* z  = (const float*)d_in[0];
  const float* C  = (const float*)d_in[1];
  const float* ar = (const float*)d_in[2];
  const float* W  = (const float*)d_in[3];
  const float* b  = (const float*)d_in[4];
  char*  ws  = (char*)d_ws;   // needs >= 918016 bytes
  float* out = (float*)d_out;

  vlad_prep<<<(Pn * Dn) / 256, 256, 0, stream>>>(C, W, ws);
  vlad_main<<<Bn / 128, 256, LDS_TOTAL, stream>>>(z, C, ar, b, ws, out);
  vlad_norm<<<(Bn * (Pn / 4)) / 256, 256, 0, stream>>>(out, (const float*)(ws + INV_OFF));
}
